// StackRNN_27711128993872
// MI455X (gfx1250) — matrix operand order repaired and checked by reading, not compiled, not run on hardware
//
#include <hip/hip_runtime.h>
#include <hip/hip_bf16.h>
#include <stdint.h>

// Problem dims (StackRNN): S=512, B=256, I=256, H=1024, SW=256, SD=100
#define S_  512
#define B_  256
#define I_  256
#define H_  1024
#define SW_ 256
#define SD_ 100
#define K_  1536            // I + SW + H
#define KT_GATES 48         // K_/32
#define KT_D     32         // H/32

// ---------------- workspace layout (bytes) ----------------
static constexpr size_t WB_OFF   = 0;                      // bf16 swizzled [48*64*4 tiles][32 lanes][8 dw] = 12,582,912 B
static constexpr size_t DWB_OFF  = 12582912;               // bf16 swizzled [32*16 tiles][32][8 dw]         =    524,288 B
static constexpr size_t ABUF_OFF = 13107200;               // bf16 swizzled [16*48 tiles][32][8 dw]         =    786,432 B
static constexpr size_t HBUF_OFF = 13893632;               // f32  [B][H]                                   =  1,048,576 B
static constexpr size_t CBUF_OFF = 14942208;               // f32  [B][H]                                   =  1,048,576 B
static constexpr size_t DBUF_OFF = 15990784;               // f32  [B][SW]                                  =    262,144 B
static constexpr size_t CTRL_OFF = 16252928;               // f32  [B][3] (+pad)                            =      4,096 B
static constexpr size_t STKA_OFF = 16257024;               // f32  [B][SD][SW]                              = 26,214,400 B
static constexpr size_t STKB_OFF = 42471424;               // f32  [B][SD][SW]                              = 26,214,400 B

// ---------------- WMMA types ----------------
typedef __attribute__((ext_vector_type(16))) __bf16 v16bf;
typedef __attribute__((ext_vector_type(8)))  float  v8f;
union Frag { v16bf v; unsigned u[8]; };

__device__ __forceinline__ unsigned bf16_rne(float f) {
    unsigned u = __float_as_uint(f);
    return (u + 0x7FFFu + ((u >> 16) & 1u)) >> 16;
}
__device__ __forceinline__ unsigned pack2(float lo, float hi) {
    return bf16_rne(lo) | (bf16_rne(hi) << 16);
}
__device__ __forceinline__ float sigm(float x) { return 1.0f / (1.0f + __expf(-x)); }

// ---------------------------------------------------------------------------
// B-matrix lane layout for v_wmma_f32_16x16x32_bf16 (32x16, K x N):
//   lane L: N = L&15, g = L>>4; VGPR v holds K = 32*kt + 16*(v>>2) + 8*g + 2*(v&3), +1 (packed bf16)
// Tiles stored gate-innermost: tile = (kt*64 + nt)*4 + G, each [32 lanes][8 dw]
// so one wave reads its 4 gate fragments from one base + immediate offsets.
// ---------------------------------------------------------------------------
__global__ void prep_wb(const float* __restrict__ W_ih, const float* __restrict__ W_hh,
                        unsigned* __restrict__ Wb) {
    int tid  = blockIdx.x * 256 + threadIdx.x;       // 48*64*4*32 = 393216 threads
    int lane = tid & 31;
    int tile = tid >> 5;                              // (kt*64 + nt)*4 + G
    int G    = tile & 3;
    int nt   = (tile >> 2) & 63;
    int kt   = tile >> 8;
    int g = lane >> 4, l16 = lane & 15;
    int n = G * H_ + nt * 16 + l16;                   // gate-major column 0..4095
    unsigned* outp = Wb + (size_t)tile * 256 + lane * 8;
#pragma unroll
    for (int v = 0; v < 8; ++v) {
        int k = kt * 32 + ((v & 4) ? 16 : 0) + 8 * g + 2 * (v & 3);             // even; pair stays on one side of 512
        float f0, f1;
        if (k < 512) { f0 = W_ih[(size_t)n * 512 + k];        f1 = W_ih[(size_t)n * 512 + k + 1]; }
        else         { f0 = W_hh[(size_t)n * 1024 + (k-512)]; f1 = W_hh[(size_t)n * 1024 + (k-511)]; }
        outp[v] = pack2(f0, f1);
    }
}

__global__ void prep_dwb(const float* __restrict__ D_w, unsigned* __restrict__ Dwb) {
    int tid  = blockIdx.x * 256 + threadIdx.x;       // 32*16*32 = 16384 threads
    int lane = tid & 31;
    int tile = tid >> 5;                              // kt*16 + nt
    int nt   = tile & 15, kt = tile >> 4;
    int g = lane >> 4, l16 = lane & 15;
    int n = nt * 16 + l16;                            // 0..255
    unsigned* outp = Dwb + (size_t)tile * 256 + lane * 8;
#pragma unroll
    for (int v = 0; v < 8; ++v) {
        int k = kt * 32 + ((v & 4) ? 16 : 0) + 8 * g + 2 * (v & 3);             // 0..1023
        outp[v] = pack2(D_w[(size_t)n * 1024 + k], D_w[(size_t)n * 1024 + k + 1]);
    }
}

// ---------------- per-step: stage A = [x_t | stack_top | h] pre-swizzled bf16 ----------------
// A layout (16-bit A 16x32): lanes 0-15 hold K{0-7,16-23}, lanes 16-31 hold K{8-15,24-31}.
// Stored per-tile lane-contiguous: Asw[(mt*48+kt)*256 + lane*8 + v]
__global__ void assemble_A(const float* __restrict__ x_t, const float* __restrict__ stackCur,
                           const float* __restrict__ hbuf, unsigned* __restrict__ Asw) {
    int idx  = blockIdx.x * 256 + threadIdx.x;       // 16*48*32*8 = 196608 dwords
    int v    = idx & 7;
    int lane = (idx >> 3) & 31;
    int tile = idx >> 8;                              // mt*48 + kt
    int mt = tile / KT_GATES, kt = tile - mt * KT_GATES;
    int g = lane >> 4, l16 = lane & 15;
    int m = mt * 16 + l16;
    int kk = kt * 32 + ((v & 4) ? 16 : 0) + 8 * g + 2 * (v & 3);  // even pair base
    float f0, f1;
    if (kk < I_) {
        f0 = x_t[m * I_ + kk]; f1 = x_t[m * I_ + kk + 1];
    } else if (kk < I_ + SW_) {
        const float* sp = stackCur + (size_t)m * (SD_ * SW_) + (kk - I_);   // stack[b][0][:]
        f0 = sp[0]; f1 = sp[1];
    } else {
        const float* hp = hbuf + (size_t)m * H_ + (kk - I_ - SW_);
        f0 = hp[0]; f1 = hp[1];
    }
    Asw[idx] = pack2(f0, f1);
}

// ---------------- per-step: gates GEMM + fused LSTM cell ----------------
// grid (16, 64), 32 threads (one wave): 16 batch rows x 16 H-cols x 4 gates.
// Double-buffered register pipeline: loads for kt+1 issue before WMMAs for kt.
__global__ void gates_gemm_cell(const unsigned* __restrict__ Asw,
                                const unsigned* __restrict__ Wb,
                                const float* __restrict__ b_ih, const float* __restrict__ b_hh,
                                float* __restrict__ hbuf, float* __restrict__ cbuf,
                                float* __restrict__ outs_t) {
    const int lane = threadIdx.x;
    const int mt = blockIdx.x;                        // 0..15 (batch tiles)
    const int nt = blockIdx.y;                        // 0..63 (H tiles)
    const int g = lane >> 4, l16 = lane & 15;

    const unsigned* ap = Asw + ((size_t)mt * KT_GATES) * 256 + lane * 8;   // +256 dw per kt
    const unsigned* bp = Wb  + ((size_t)nt * 4) * 256 + lane * 8;          // +65536 dw per kt; gate at +G*256

    v8f acc[4] = {};
    Frag A2[2];
    Frag Bf[2][4];

    // preload kt = 0
#pragma unroll
    for (int v = 0; v < 8; ++v) A2[0].u[v] = ap[v];
#pragma unroll
    for (int G = 0; G < 4; ++G)
#pragma unroll
        for (int v = 0; v < 8; ++v) Bf[0][G].u[v] = bp[G * 256 + v];

#pragma unroll 2
    for (int kt = 0; kt < KT_GATES; ++kt) {
        const int cur = kt & 1, nxt = cur ^ 1;
        // issue next-slab loads first (kt=47 issues a discarded in-workspace overread)
        const unsigned* apn = ap + (size_t)(kt + 1) * 256;
        const unsigned* bpn = bp + (size_t)(kt + 1) * 65536;
#pragma unroll
        for (int v = 0; v < 8; ++v) A2[nxt].u[v] = apn[v];
#pragma unroll
        for (int G = 0; G < 4; ++G)
#pragma unroll
            for (int v = 0; v < 8; ++v) Bf[nxt][G].u[v] = bpn[G * 256 + v];
        // compute current slab while next loads are in flight
#pragma unroll
        for (int G = 0; G < 4; ++G)
            acc[G] = __builtin_amdgcn_wmma_f32_16x16x32_bf16(
                false, A2[cur].v, false, Bf[cur][G].v, (short)0, acc[G], false, false);
    }

    const int n = nt * 16 + l16;                      // C/D layout: lane = N column
    float bi0 = b_ih[n]          + b_hh[n];
    float bi1 = b_ih[H_ + n]     + b_hh[H_ + n];
    float bi2 = b_ih[2 * H_ + n] + b_hh[2 * H_ + n];
    float bi3 = b_ih[3 * H_ + n] + b_hh[3 * H_ + n];
#pragma unroll
    for (int r = 0; r < 8; ++r) {                     // C/D layout: row M = 8*g + r
        int row = mt * 16 + 8 * g + r;
        float iv = sigm(acc[0][r] + bi0);
        float fv = sigm(acc[1][r] + bi1);
        float gv = tanhf(acc[2][r] + bi2);
        float ov = sigm(acc[3][r] + bi3);
        size_t off = (size_t)row * H_ + n;
        float cn = fv * cbuf[off] + iv * gv;
        float hn = ov * tanhf(cn);
        cbuf[off] = cn;
        hbuf[off] = hn;
        outs_t[off] = hn;
    }
}

// ---------------- per-step: controls = softmax(h @ A_w^T + A_b) ----------------
__global__ void controls_kernel(const float* __restrict__ hbuf, const float* __restrict__ A_w,
                                const float* __restrict__ A_b, float* __restrict__ ctrl) {
    int b = blockIdx.x, lane = threadIdx.x;
    float s0 = 0.f, s1 = 0.f, s2 = 0.f;
    for (int k = lane; k < H_; k += 32) {
        float hv = hbuf[b * H_ + k];
        s0 += hv * A_w[k]; s1 += hv * A_w[H_ + k]; s2 += hv * A_w[2 * H_ + k];
    }
#pragma unroll
    for (int off = 16; off > 0; off >>= 1) {
        s0 += __shfl_down(s0, off, 32);
        s1 += __shfl_down(s1, off, 32);
        s2 += __shfl_down(s2, off, 32);
    }
    if (lane == 0) {
        s0 += A_b[0]; s1 += A_b[1]; s2 += A_b[2];
        float mx = fmaxf(s0, fmaxf(s1, s2));
        float e0 = __expf(s0 - mx), e1 = __expf(s1 - mx), e2 = __expf(s2 - mx);
        float inv = 1.0f / (e0 + e1 + e2);
        ctrl[b * 3 + 0] = e0 * inv; ctrl[b * 3 + 1] = e1 * inv; ctrl[b * 3 + 2] = e2 * inv;
    }
}

// ---------------- per-step: d = tanh(h @ D_w^T + D_b) via WMMA (double-buffered) ----------------
__global__ void d_gemm(const float* __restrict__ hbuf, const unsigned* __restrict__ Dwb,
                       const float* __restrict__ D_b, float* __restrict__ dbuf) {
    int lane = threadIdx.x, mt = blockIdx.x, nt = blockIdx.y;   // (16, 16)
    int g = lane >> 4, l16 = lane & 15, m = mt * 16 + l16;
    const float*    hb = hbuf + (size_t)m * H_ + 8 * g;          // +32 floats per kt
    const unsigned* bp = Dwb + (size_t)nt * 256 + lane * 8;      // +4096 dw per kt

    v8f acc = {};
    Frag A2[2], Bf[2];

    auto loadA = [&](Frag& a, const float* h) {
#pragma unroll
        for (int v = 0; v < 8; ++v) {
            int off = ((v & 4) ? 16 : 0) + 2 * (v & 3);          // + k within slab (g folded into hb)
            a.u[v] = pack2(h[off], h[off + 1]);
        }
    };

    loadA(A2[0], hb);
#pragma unroll
    for (int v = 0; v < 8; ++v) Bf[0].u[v] = bp[v];

#pragma unroll 2
    for (int kt = 0; kt < KT_D; ++kt) {
        int cur = kt & 1, nxt = cur ^ 1;
        loadA(A2[nxt], hb + (size_t)(kt + 1) * 32);              // kt=31: in-workspace overread, discarded
        const unsigned* bpn = bp + (size_t)(kt + 1) * 4096;
#pragma unroll
        for (int v = 0; v < 8; ++v) Bf[nxt].u[v] = bpn[v];
        acc = __builtin_amdgcn_wmma_f32_16x16x32_bf16(
            false, A2[cur].v, false, Bf[cur].v, (short)0, acc, false, false);
    }
    int n = nt * 16 + l16;
    float db = D_b[n];
#pragma unroll
    for (int r = 0; r < 8; ++r) {
        int row = mt * 16 + 8 * g + r;
        dbuf[(size_t)row * SW_ + n] = tanhf(acc[r] + db);
    }
}

// ---------------- per-step: stack shift/blend (ping-pong, L2-resident) ----------------
__global__ void stack_update(const float* __restrict__ cur, const float* __restrict__ dbuf,
                             const float* __restrict__ ctrl, float* __restrict__ nxt) {
    int idx = blockIdx.x * 256 + threadIdx.x;        // B_*SD_*SW_ = 6,553,600
    int w  = idx & (SW_ - 1);
    int bs = idx >> 8;                                // b*SD_ + s
    int s  = bs % SD_, b = bs / SD_;
    float a_push = ctrl[b * 3 + 0], a_pop = ctrl[b * 3 + 1], a_noop = ctrl[b * 3 + 2];
    float up = (s == 0)       ? dbuf[b * SW_ + w] : cur[idx - SW_];
    float dn = (s == SD_ - 1) ? 0.0f              : cur[idx + SW_];
    nxt[idx] = a_noop * cur[idx] + a_push * up + a_pop * dn;
}

// ---------------- host ----------------
extern "C" void kernel_launch(void* const* d_in, const int* in_sizes, int n_in,
                              void* d_out, int out_size, void* d_ws, size_t ws_size,
                              hipStream_t stream) {
    const float* x      = (const float*)d_in[0];
    const float* h0     = (const float*)d_in[1];
    const float* c0     = (const float*)d_in[2];
    const float* stack0 = (const float*)d_in[3];
    const float* W_ih   = (const float*)d_in[4];
    const float* W_hh   = (const float*)d_in[5];
    const float* b_ih   = (const float*)d_in[6];
    const float* b_hh   = (const float*)d_in[7];
    const float* A_w    = (const float*)d_in[8];
    const float* A_b    = (const float*)d_in[9];
    const float* D_w    = (const float*)d_in[10];
    const float* D_b    = (const float*)d_in[11];
    float* out = (float*)d_out;
    char*  ws  = (char*)d_ws;

    unsigned* Wb   = (unsigned*)(ws + WB_OFF);
    unsigned* Dwb  = (unsigned*)(ws + DWB_OFF);
    unsigned* Asw  = (unsigned*)(ws + ABUF_OFF);
    float* hbuf = (float*)(ws + HBUF_OFF);
    float* cbuf = (float*)(ws + CBUF_OFF);
    float* dbuf = (float*)(ws + DBUF_OFF);
    float* ctrl = (float*)(ws + CTRL_OFF);
    float* stk[2] = { (float*)(ws + STKA_OFF), (float*)(ws + STKB_OFF) };

    // init recurrent state (graph-capture-safe d2d copies)
    hipMemcpyAsync(hbuf,   h0,     (size_t)B_ * H_ * 4,        hipMemcpyDeviceToDevice, stream);
    hipMemcpyAsync(cbuf,   c0,     (size_t)B_ * H_ * 4,        hipMemcpyDeviceToDevice, stream);
    hipMemcpyAsync(stk[0], stack0, (size_t)B_ * SD_ * SW_ * 4, hipMemcpyDeviceToDevice, stream);

    // one-time weight pre-swizzle into WMMA-native bf16 tiles
    prep_wb <<<1536, 256, 0, stream>>>(W_ih, W_hh, Wb);
    prep_dwb<<<  64, 256, 0, stream>>>(D_w, Dwb);

    for (int t = 0; t < S_; ++t) {
        const float* x_t = x + (size_t)t * B_ * I_;
        float* curS = stk[t & 1];
        float* nxtS = stk[(t + 1) & 1];
        assemble_A     <<<768, 256, 0, stream>>>(x_t, curS, hbuf, Asw);
        gates_gemm_cell<<<dim3(16, 64), 32, 0, stream>>>(Asw, Wb, b_ih, b_hh, hbuf, cbuf,
                                                         out + (size_t)t * B_ * H_);
        controls_kernel<<<B_, 32, 0, stream>>>(hbuf, A_w, A_b, ctrl);
        d_gemm         <<<dim3(16, 16), 32, 0, stream>>>(hbuf, Dwb, D_b, dbuf);
        stack_update   <<<(B_ * SD_ * SW_) / 256, 256, 0, stream>>>(curS, dbuf, ctrl, nxtS);
    }

    // final state outputs: outs | hn | cn | stackn
    size_t outsN = (size_t)S_ * B_ * H_;
    hipMemcpyAsync(out + outsN,                   hbuf,   (size_t)B_ * H_ * 4,        hipMemcpyDeviceToDevice, stream);
    hipMemcpyAsync(out + outsN + (size_t)B_ * H_, cbuf,   (size_t)B_ * H_ * 4,        hipMemcpyDeviceToDevice, stream);
    hipMemcpyAsync(out + outsN + 2u * B_ * H_,    stk[0], (size_t)B_ * SD_ * SW_ * 4, hipMemcpyDeviceToDevice, stream);
}
